// ScaledDotProductAttentionMemory_62405874811701
// MI455X (gfx1250) — compile-verified, matrix operand order checked by reading, not run on hardware
//
#include <hip/hip_runtime.h>
#include <hip/hip_bf16.h>

// ---------------- types ----------------
typedef __bf16 bf16_t;
typedef bf16_t v16bf __attribute__((ext_vector_type(16)));
typedef float  v8f   __attribute__((ext_vector_type(8)));
typedef unsigned short u16x8  __attribute__((ext_vector_type(8)));
typedef unsigned short u16x16 __attribute__((ext_vector_type(16)));
typedef unsigned int   u32x4  __attribute__((ext_vector_type(4)));
typedef int            i32x8  __attribute__((ext_vector_type(8)));

#define H_HEADS 16
#define NQ 1024
#define NK 1024
#define DMODEL 1024
#define DK 64
#define NKV 1064          // nk + M
#define NKV_PAD 1088      // padded to 34 * 32
#define SST 1092          // LDS row stride (pad vs 64-bank conflicts)
#define QSTAGE_BYTES 4096 // 32 x 64 bf16 Q tile staged by TDM

__device__ __forceinline__ unsigned short f32_to_bf16(float f) {
    unsigned int u = __float_as_uint(f);
    unsigned int r = u + 0x7FFFu + ((u >> 16) & 1u);
    return (unsigned short)(r >> 16);
}

__device__ __forceinline__ v16bf make_frag(u16x8 lo, u16x8 hi) {
    u16x16 u;
#pragma unroll
    for (int i = 0; i < 8; ++i) { u[i] = lo[i]; u[i + 8] = hi[i]; }
    return __builtin_bit_cast(v16bf, u);
}

// Fragment of a 16x32 bf16 tile: rowBase points at this lane's row (stride = K).
// Lanes 0-15: K {0..7,16..23}; lanes 16-31: K {8..15,24..31} (ISA 16-bit layout).
__device__ __forceinline__ v16bf load_frag(const unsigned short* rowBase, int kChunk, int lane) {
    const int sel = ((lane >> 4) & 1) << 3;
    const u16x8 lo = *reinterpret_cast<const u16x8*>(rowBase + kChunk + sel);
    const u16x8 hi = *reinterpret_cast<const u16x8*>(rowBase + kChunk + sel + 16);
    return make_frag(lo, hi);
}

__device__ __forceinline__ v8f wmma_bf16(v16bf a, v16bf b, v8f c) {
    return __builtin_amdgcn_wmma_f32_16x16x32_bf16(false, a, false, b, (short)0, c, false, false);
}

// ---------------- f32 -> bf16 convert ----------------
__global__ __launch_bounds__(256) void convert_bf16_kernel(const float* __restrict__ in,
                                                           unsigned short* __restrict__ out, int n) {
    int i = blockIdx.x * blockDim.x + threadIdx.x;
    int stride = gridDim.x * blockDim.x;
    for (; i < n; i += stride) out[i] = f32_to_bf16(in[i]);
}

// ---------------- W (KxN f32) -> Wt (NxK bf16) ----------------
__global__ __launch_bounds__(256) void transpose_bf16_kernel(const float* __restrict__ W,
                                                             unsigned short* __restrict__ Wt) {
    __shared__ float tile[32][33];
    const int k0 = blockIdx.y * 32, n0 = blockIdx.x * 32;
    const int tx = threadIdx.x & 31, ty = threadIdx.x >> 5;
    for (int i = ty; i < 32; i += 8) tile[i][tx] = W[(size_t)(k0 + i) * DMODEL + n0 + tx];
    __syncthreads();
    for (int i = ty; i < 32; i += 8) Wt[(size_t)(n0 + i) * DMODEL + k0 + tx] = f32_to_bf16(tile[tx][i]);
}

// ---------------- memory slots + padding into K / Vt ----------------
__global__ __launch_bounds__(256) void slots_kernel(const float* __restrict__ m_k,
                                                    const float* __restrict__ m_v,
                                                    unsigned short* __restrict__ kws,
                                                    unsigned short* __restrict__ vtws) {
    int idx = blockIdx.x * blockDim.x + threadIdx.x;       // b*h*64*64
    if (idx >= 8 * H_HEADS * 64 * 64) return;
    const int d  = idx & 63;
    const int j  = (idx >> 6) & 63;                        // slot (pad beyond 40)
    const int hh = (idx >> 12) & (H_HEADS - 1);
    const int bb = idx >> 16;
    float kv = 0.f, vv = 0.f;
    if (j < 40) {
        kv = 8.0f        * m_k[(size_t)j * DMODEL + hh * DK + d];   // sqrt(dk)=8 pre-scale
        vv = 6.32455532f * m_v[(size_t)j * DMODEL + hh * DK + d];   // sqrt(M)=sqrt(40)
    }
    const int bh = bb * H_HEADS + hh;
    kws [(((size_t)bh * NKV_PAD) + NK + j) * DK + d]      = f32_to_bf16(kv);
    vtws[(((size_t)bh * DK)      + d) * NKV_PAD + NK + j] = f32_to_bf16(vv);
}

// ---------------- WMMA GEMM, 32x64 register tile per wave (2 A-frags x 4 B-frags)
// C(M x 1024) = A(bf16,MxK) @ Bt(bf16,NxK)^T + bias.  Block: 8 waves as 4(M) x 2(N)
// -> 128x128 block tile.  grid = (N/128, M/128).
// mode 0: Q -> [b,h,tok,dk] bf16   mode 1: K -> [b,h(1088 rows),tok,dk] bf16
// mode 2: V -> [b,h,dk,1088] bf16  mode 3: f32 out (final projection) to d_out
__global__ __launch_bounds__(256) void gemm_bf16_wmma(const unsigned short* __restrict__ A,
                                                      const unsigned short* __restrict__ Bt,
                                                      const float* __restrict__ bias,
                                                      void* __restrict__ out,
                                                      int K, int mode, int rowsPad) {
    const int lane = threadIdx.x & 31;
    const int w    = threadIdx.x >> 5;
    const int rowBase = blockIdx.y * 128 + (w >> 1) * 32;
    const int colBase = blockIdx.x * 128 + (w & 1) * 64;

    const unsigned short* aRow0 = A + (size_t)(rowBase + (lane & 15)) * K;
    const unsigned short* aRow1 = aRow0 + (size_t)16 * K;
    const unsigned short* bRow0 = Bt + (size_t)(colBase +  0 + (lane & 15)) * K;
    const unsigned short* bRow1 = bRow0 + (size_t)16 * K;
    const unsigned short* bRow2 = bRow0 + (size_t)32 * K;
    const unsigned short* bRow3 = bRow0 + (size_t)48 * K;

    v8f acc[2][4] = {};
    for (int k0 = 0; k0 < K; k0 += 32) {
        __builtin_prefetch(aRow0 + k0 + 256, 0, 1);
        const v16bf a0 = load_frag(aRow0, k0, lane);
        const v16bf a1 = load_frag(aRow1, k0, lane);
        const v16bf b0 = load_frag(bRow0, k0, lane);
        const v16bf b1 = load_frag(bRow1, k0, lane);
        const v16bf b2 = load_frag(bRow2, k0, lane);
        const v16bf b3 = load_frag(bRow3, k0, lane);
        acc[0][0] = wmma_bf16(a0, b0, acc[0][0]);
        acc[1][0] = wmma_bf16(a1, b0, acc[1][0]);
        acc[0][1] = wmma_bf16(a0, b1, acc[0][1]);
        acc[1][1] = wmma_bf16(a1, b1, acc[1][1]);
        acc[0][2] = wmma_bf16(a0, b2, acc[0][2]);
        acc[1][2] = wmma_bf16(a1, b2, acc[1][2]);
        acc[0][3] = wmma_bf16(a0, b3, acc[0][3]);
        acc[1][3] = wmma_bf16(a1, b3, acc[1][3]);
    }

    const int nl   = (lane < 16) ? lane : lane - 16;
    const int mAdd = (lane < 16) ? 0 : 8;
#pragma unroll
    for (int mi = 0; mi < 2; ++mi) {
#pragma unroll
        for (int t = 0; t < 4; ++t) {
#pragma unroll
            for (int r = 0; r < 8; ++r) {
                const int m = rowBase + mi * 16 + r + mAdd;
                const int n = colBase + t * 16 + nl;
                const float val = acc[mi][t][r] + bias[n];
                if (mode == 3) {
                    ((float*)out)[(size_t)m * DMODEL + n] = val;
                } else {
                    const int bb = m >> 10, tok = m & 1023, hh = n >> 6, d = n & 63;
                    const unsigned short bvv = f32_to_bf16(val);
                    unsigned short* o = (unsigned short*)out;
                    if (mode <= 1)      o[(((size_t)(bb * H_HEADS + hh) * rowsPad) + tok) * DK + d] = bvv;
                    else                o[(((size_t)(bb * H_HEADS + hh) * DK) + d) * NKV_PAD + tok] = bvv;
                }
            }
        }
    }
}

// ---------------- fused attention ----------------
// TDM stages the 32x64 Q tile into LDS; scores via WMMA into a 32x1092 f32 LDS
// tile (needs CDNA5 320KB LDS); masked softmax in LDS; att streamed out; PV WMMA.
__global__ __launch_bounds__(256) void attn_kernel(const unsigned short* __restrict__ qws,
                                                   const unsigned short* __restrict__ kws,
                                                   const unsigned short* __restrict__ vtws,
                                                   const unsigned char* __restrict__ mask,
                                                   float* __restrict__ att_out,
                                                   unsigned short* __restrict__ ows) {
    extern __shared__ char smem[];
    unsigned short* qtile = (unsigned short*)smem;                  // QSTAGE_BYTES
    float* sS   = (float*)(smem + QSTAGE_BYTES);                    // 32 x SST
    float* sRed = sS + 32 * SST;                                    // 32 x 8
    const int bh = blockIdx.x;                  // b*H + h
    const int bb = bh >> 4, hh = bh & (H_HEADS - 1);
    const int qt = blockIdx.y;                  // 32-row query tile
    const int tid = threadIdx.x, lane = tid & 31, w = tid >> 5;
    const int rowHalf = w & 1;

    const unsigned short* qb = qws  + (size_t)bh * NQ * DK;
    const unsigned short* kb = kws  + (size_t)bh * NKV_PAD * DK;
    const unsigned short* vb = vtws + (size_t)bh * DK * NKV_PAD;

    // ---- TDM: stage contiguous 32x64 bf16 Q tile (4KB) into LDS ----
    if (w == 0) {
        const unsigned long long ga =
            (unsigned long long)(const void*)(qb + (size_t)(qt * 32) * DK);
        u32x4 g0;
        g0[0] = 1u;                                            // count=1, user mode
        g0[1] = 0u;                                            // lds_addr = 0 (qtile)
        g0[2] = (unsigned)(ga & 0xFFFFFFFFu);                  // global_addr[31:0]
        g0[3] = (unsigned)((ga >> 32) & 0x01FFFFFFu) | (2u << 30);  // addr[56:32] | type=2
        const unsigned td0 = 2048;                             // elements (32*64)
        i32x8 g1;
        g1[0] = (int)(1u << 16);                               // data_size=2B, no flags
        g1[1] = (int)((td0 & 0xFFFFu) << 16);                  // tensor_dim0[15:0]
        g1[2] = (int)((td0 >> 16) | (1u << 16));               // tensor_dim0[31:16], tensor_dim1=1
        g1[3] = (int)(td0 << 16);                              // tile_dim0 = 2048
        g1[4] = 1;                                             // tile_dim1=1, tile_dim2=0
        g1[5] = (int)td0;                                      // tensor_dim0_stride
        g1[6] = 0;
        g1[7] = 0;
        asm volatile("tensor_load_to_lds %0, %1" :: "s"(g0), "s"(g1) : "memory");
        __builtin_amdgcn_s_wait_tensorcnt(0);
    }
    __syncthreads();

    // ---- phase 1: S = (Q Kt) / sqrt(dk), Q fragments from LDS ----
    {
        const unsigned short* qr = qtile + (size_t)(rowHalf * 16 + (lane & 15)) * DK;
        const v16bf a0 = load_frag(qr, 0, lane);
        const v16bf a1 = load_frag(qr, 32, lane);
        const int nl = (lane < 16) ? lane : lane - 16;
        const int mAdd = (lane < 16) ? 0 : 8;
        for (int ct = (w >> 1); ct < NKV_PAD / 16; ct += 4) {
            const unsigned short* kr = kb + (size_t)(ct * 16 + (lane & 15)) * DK;
            v8f c = {};
            c = wmma_bf16(a0, load_frag(kr, 0, lane), c);
            c = wmma_bf16(a1, load_frag(kr, 32, lane), c);
#pragma unroll
            for (int r = 0; r < 8; ++r)
                sS[(rowHalf * 16 + r + mAdd) * SST + ct * 16 + nl] = c[r] * 0.125f;
        }
    }
    __syncthreads();

    // ---- phase 2: mask + softmax, write att output ----
    const int row = tid >> 3, g = tid & 7;          // 8 threads per query row
    const int qIdx = qt * 32 + row;
    float* Sr = sS + row * SST;
    const unsigned char* mrow = mask + ((size_t)bh * NQ + qIdx) * NK;
    const float NEG = -__builtin_inff();
    float mx = -3.4e38f;
    for (int j = g * 136; j < g * 136 + 136; ++j) {
        float x = Sr[j];
        if (j >= NKV) x = NEG;                       // padding
        else if (j < NK && mrow[j]) x = NEG;         // masked real keys; slots always live
        Sr[j] = x;
        mx = fmaxf(mx, x);
    }
    sRed[row * 8 + g] = mx;
    __syncthreads();
    float rmx = sRed[row * 8];
#pragma unroll
    for (int i = 1; i < 8; ++i) rmx = fmaxf(rmx, sRed[row * 8 + i]);
    float sum = 0.f;
    for (int j = g * 136; j < g * 136 + 136; ++j) {
        float e = __expf(Sr[j] - rmx);
        Sr[j] = e;
        sum += e;
    }
    __syncthreads();
    sRed[row * 8 + g] = sum;
    __syncthreads();
    float tot = 0.f;
#pragma unroll
    for (int i = 0; i < 8; ++i) tot += sRed[row * 8 + i];
    const float inv = 1.0f / tot;
    float* attRow = att_out + ((size_t)bh * NQ + qIdx) * NKV;
    for (int j = g * 136; j < g * 136 + 136; ++j) {
        float p = Sr[j] * inv;
        Sr[j] = p;
        if (j < NKV) attRow[j] = p;
    }
    __syncthreads();

    // ---- phase 3: O = P @ V  (P from LDS f32 -> bf16, V transposed, contiguous) ----
    {
        const int dt = w >> 1;                       // 16-wide dv tile
        const int aRowL = rowHalf * 16 + (lane & 15);
        const unsigned short* vr = vb + (size_t)(dt * 16 + (lane & 15)) * NKV_PAD;
        v8f o = {};
        for (int ks = 0; ks < NKV_PAD / 32; ++ks) {
            const int kv0 = ks * 32;
            u16x16 u;
            const float* ar = sS + (size_t)aRowL * SST + kv0 + (((lane >> 4) & 1) << 3);
#pragma unroll
            for (int i = 0; i < 8; ++i) { u[i] = f32_to_bf16(ar[i]); u[i + 8] = f32_to_bf16(ar[16 + i]); }
            v16bf a = __builtin_bit_cast(v16bf, u);
            o = wmma_bf16(a, load_frag(vr, kv0, lane), o);
        }
        const int nl = (lane < 16) ? lane : lane - 16;
        const int mAdd = (lane < 16) ? 0 : 8;
#pragma unroll
        for (int r = 0; r < 8; ++r) {
            const int qI = qt * 32 + rowHalf * 16 + r + mAdd;
            ows[((size_t)bb * NQ + qI) * DMODEL + hh * DK + dt * 16 + nl] = f32_to_bf16(o[r]);
        }
    }
}

// ---------------- host launcher ----------------
extern "C" void kernel_launch(void* const* d_in, const int* in_sizes, int n_in,
                              void* d_out, int out_size, void* d_ws, size_t ws_size,
                              hipStream_t stream) {
    const float* queries = (const float*)d_in[0];
    const float* keys    = (const float*)d_in[1];
    const float* values  = (const float*)d_in[2];
    const unsigned char* mask = (const unsigned char*)d_in[3];   // numpy bool = 1 byte
    const float* Wq = (const float*)d_in[4];
    const float* bq = (const float*)d_in[5];
    const float* Wk = (const float*)d_in[6];
    const float* bk = (const float*)d_in[7];
    const float* Wv = (const float*)d_in[8];
    const float* bv = (const float*)d_in[9];
    const float* Wo = (const float*)d_in[10];
    const float* bo = (const float*)d_in[11];
    const float* m_k = (const float*)d_in[12];
    const float* m_v = (const float*)d_in[13];

    char* ws = (char*)d_ws;
    size_t off = 0;
    auto alloc = [&](size_t bytes) -> void* {
        void* p = ws + off;
        off += (bytes + 255) & ~(size_t)255;
        return p;
    };
    const size_t MROWS = 8 * (size_t)NQ;          // 8192
    unsigned short* xq  = (unsigned short*)alloc(MROWS * DMODEL * 2);
    unsigned short* xk  = (unsigned short*)alloc(MROWS * DMODEL * 2);
    unsigned short* xv  = (unsigned short*)alloc(MROWS * DMODEL * 2);
    unsigned short* wqt = (unsigned short*)alloc((size_t)DMODEL * DMODEL * 2);
    unsigned short* wkt = (unsigned short*)alloc((size_t)DMODEL * DMODEL * 2);
    unsigned short* wvt = (unsigned short*)alloc((size_t)DMODEL * DMODEL * 2);
    unsigned short* wot = (unsigned short*)alloc((size_t)DMODEL * DMODEL * 2);
    unsigned short* qws = (unsigned short*)alloc((size_t)8 * H_HEADS * NQ * DK * 2);
    unsigned short* kws = (unsigned short*)alloc((size_t)8 * H_HEADS * NKV_PAD * DK * 2);
    unsigned short* vtw = (unsigned short*)alloc((size_t)8 * H_HEADS * DK * NKV_PAD * 2);
    unsigned short* ows = (unsigned short*)alloc(MROWS * DMODEL * 2);
    (void)ws_size; (void)in_sizes; (void)n_in; (void)out_size;

    float* out_f = (float*)d_out;                            // (8,1024,1024)
    float* att_f = out_f + (size_t)8 * NQ * DMODEL;          // (8,16,1024,1064)

    const dim3 tb(256);
    const int nElem = (int)(MROWS * DMODEL);
    convert_bf16_kernel<<<dim3(2048), tb, 0, stream>>>(queries, xq, nElem);
    convert_bf16_kernel<<<dim3(2048), tb, 0, stream>>>(keys,    xk, nElem);
    convert_bf16_kernel<<<dim3(2048), tb, 0, stream>>>(values,  xv, nElem);
    transpose_bf16_kernel<<<dim3(32, 32), tb, 0, stream>>>(Wq, wqt);
    transpose_bf16_kernel<<<dim3(32, 32), tb, 0, stream>>>(Wk, wkt);
    transpose_bf16_kernel<<<dim3(32, 32), tb, 0, stream>>>(Wv, wvt);
    transpose_bf16_kernel<<<dim3(32, 32), tb, 0, stream>>>(Wo, wot);

    gemm_bf16_wmma<<<dim3(8, 64), tb, 0, stream>>>(xq, wqt, bq, qws, DMODEL, 0, NQ);
    gemm_bf16_wmma<<<dim3(8, 64), tb, 0, stream>>>(xk, wkt, bk, kws, DMODEL, 1, NKV_PAD);
    gemm_bf16_wmma<<<dim3(8, 64), tb, 0, stream>>>(xv, wvt, bv, vtw, DMODEL, 2, NKV_PAD);
    slots_kernel<<<dim3(2048), tb, 0, stream>>>(m_k, m_v, kws, vtw);

    const size_t smemBytes = QSTAGE_BYTES + (size_t)(32 * SST + 32 * 8) * sizeof(float); // ~145 KB
    attn_kernel<<<dim3(8 * H_HEADS, NQ / 32), tb, smemBytes, stream>>>(qws, kws, vtw, mask, att_f, ows);

    gemm_bf16_wmma<<<dim3(8, 64), tb, 0, stream>>>(ows, wot, bo, out_f, DMODEL, 3, 0);
}
